// HierarchicalCSNet_19628000542751
// MI455X (gfx1250) — matrix operand order repaired and checked by reading, not compiled, not run on hardware
//
#include <hip/hip_runtime.h>
#include <hip/hip_bf16.h>

typedef __attribute__((ext_vector_type(16))) _Float16 v16h;
typedef __attribute__((ext_vector_type(8)))  _Float16 v8h;
typedef __attribute__((ext_vector_type(8)))  float    v8f;

#define HH 256
#define WW 256
#define NB 8

#ifndef __has_builtin
#define __has_builtin(x) 0
#endif

#if defined(__HIP_DEVICE_COMPILE__) && __has_builtin(__builtin_amdgcn_global_load_async_to_lds_b128)
#define USE_ASYNC_LDS 1
#else
#define USE_ASYNC_LDS 0
#endif

#if USE_ASYNC_LDS
// Exact pointee type from the builtin's signature: int __attribute__((vector_size(16)))
typedef int v4i_t __attribute__((vector_size(16)));
typedef __attribute__((address_space(1))) v4i_t gv4i_t;   // global (printed as __device__)
typedef __attribute__((address_space(3))) v4i_t lv4i_t;   // LDS
__device__ static inline gv4i_t* gcast(const void* p) {
  return (gv4i_t*)(unsigned long long)p;
}
__device__ static inline lv4i_t* lcast(const void* p) {
  return (lv4i_t*)(unsigned int)(unsigned long long)p;
}
#endif

__device__ static inline void async_stage_wait() {
#if USE_ASYNC_LDS
#if __has_builtin(__builtin_amdgcn_s_wait_asynccnt)
  __builtin_amdgcn_s_wait_asynccnt(0);
#else
  asm volatile("s_wait_asynccnt 0" ::: "memory");
#endif
#endif
}

// ---------------------------------------------------------------------------
// Zero guard buffer init (for async OOB redirect)
// ---------------------------------------------------------------------------
__global__ void zero_init_kernel(_Float16* p, int n) {
  int i = blockIdx.x * blockDim.x + threadIdx.x;
  if (i < n) p[i] = (_Float16)0.f;
}

// ---------------------------------------------------------------------------
// Weight repack: f32 [Cout][Cin][3][3] -> f16 [Cout][tap][Cin]  (tap-major K)
// ---------------------------------------------------------------------------
__global__ void convert_w_kernel(const float* __restrict__ src,
                                 _Float16* __restrict__ dst,
                                 int cout, int cin) {
  int i = blockIdx.x * blockDim.x + threadIdx.x;
  int total = cout * cin * 9;
  if (i >= total) return;
  int co  = i / (cin * 9);
  int r   = i % (cin * 9);
  int ci  = r / 9;
  int tap = r % 9;
  dst[(size_t)co * cin * 9 + tap * cin + ci] = (_Float16)src[i];
}

// ---------------------------------------------------------------------------
// Head part 1: 32x32 stride-32 sampling conv. x:[8,1,256,256] -> samp:[8,13,64]
// ---------------------------------------------------------------------------
__global__ void sample_kernel(const float* __restrict__ x,
                              const float* __restrict__ sw,   // [13][1024]
                              float* __restrict__ samp) {
  int i = blockIdx.x * blockDim.x + threadIdx.x;
  if (i >= 8 * 13 * 64) return;
  int b   = i / (13 * 64);
  int r   = i % (13 * 64);
  int sc  = r / 64;
  int pos = r % 64;
  int hb = pos >> 3, wb = pos & 7;
  const float* xp = x + (size_t)(b * HH + hb * 32) * WW + wb * 32;
  const float* wp = sw + sc * 1024;
  float acc = 0.f;
  for (int ii = 0; ii < 32; ++ii)
    for (int jj = 0; jj < 32; ++jj)
      acc += xp[ii * WW + jj] * wp[ii * 32 + jj];
  samp[i] = acc;
}

// ---------------------------------------------------------------------------
// Head part 2: 1x1 up-conv (13->1024) fused with depth_to_space(32).
// samp:[8,13,64] -> img:[8,256,256] f16
// ---------------------------------------------------------------------------
__global__ void up_dts_kernel(const float* __restrict__ samp,
                              const float* __restrict__ upw,  // [1024][13]
                              const float* __restrict__ upb,  // [1024]
                              _Float16* __restrict__ img) {
  int i = blockIdx.x * blockDim.x + threadIdx.x;
  if (i >= 8 * 65536) return;
  int b = i >> 16;
  int h = (i >> 8) & 255;
  int w = i & 255;
  int ch = ((h & 31) << 5) + (w & 31);
  int hb = h >> 5, wb = w >> 5;
  const float* sp = samp + b * 13 * 64 + hb * 8 + wb;
  const float* wp = upw + ch * 13;
  float acc = upb[ch];
#pragma unroll
  for (int sc = 0; sc < 13; ++sc) acc += wp[sc] * sp[sc * 64];
  img[i] = (_Float16)acc;
}

// ---------------------------------------------------------------------------
// Head part 3: 3x3 conv 1->64 + bias + PReLU. img f16 -> head NHWC f16.
// ---------------------------------------------------------------------------
__global__ void conv1_kernel(const _Float16* __restrict__ img,
                             const float* __restrict__ w,     // [64][9]
                             const float* __restrict__ bias,  // [64]
                             const float* __restrict__ alpha, // [1]
                             _Float16* __restrict__ out) {    // [8,256,256,64]
  int i = blockIdx.x * blockDim.x + threadIdx.x;
  if (i >= 8 * 65536) return;
  int b = i >> 16;
  int h = (i >> 8) & 255;
  int wx = i & 255;
  float v[9];
#pragma unroll
  for (int tap = 0; tap < 9; ++tap) {
    int hh = h + tap / 3 - 1, ww = wx + tap % 3 - 1;
    v[tap] = (hh >= 0 && hh < HH && ww >= 0 && ww < WW)
                 ? (float)img[(size_t)(b * HH + hh) * WW + ww] : 0.f;
  }
  float a = alpha[0];
  _Float16* op = out + ((size_t)i << 6);
  for (int c8 = 0; c8 < 8; ++c8) {
    v8h pk;
#pragma unroll
    for (int j = 0; j < 8; ++j) {
      int co = c8 * 8 + j;
      const float* wp = w + co * 9;
      float acc = bias[co];
#pragma unroll
      for (int tap = 0; tap < 9; ++tap) acc += wp[tap] * v[tap];
      pk[j] = (_Float16)(acc >= 0.f ? acc : a * acc);
    }
    *(v8h*)(op + c8 * 8) = pk;
  }
}

// ---------------------------------------------------------------------------
// WMMA implicit-GEMM 3x3 conv, pad=1, Cout=64, NHWC f16 activations.
// Double-buffered LDS; staging via GLOBAL_LOAD_ASYNC_TO_LDS_B128 (ASYNCcnt).
// CIN==128: implicit concat, channels [0,64) from in0, [64,128) from in1.
// Per block: 64 cout x 64 pixels. 8 waves, 2 output tiles per wave.
// K ordered tap-major: k = tap*CIN + cin  (each 32-step stays in one tap).
// ---------------------------------------------------------------------------
template<int CIN, bool PRELU>
__global__ __launch_bounds__(256)
void conv3x3_wmma_kernel(const _Float16* __restrict__ in0,
                         const _Float16* __restrict__ in1,
                         const _Float16* __restrict__ wgt,   // [64][9*CIN] f16
                         const float* __restrict__ bias,     // [64]
                         const float* __restrict__ alpha,    // [1]
                         const _Float16* __restrict__ zbuf,  // >=8 zero halves
                         _Float16* __restrict__ out) {       // NHWC [8,256,256,64]
  constexpr int KTOT  = 9 * CIN;
  constexpr int NSTEP = KTOT / 32;

  __shared__ _Float16 As[2][64][40];   // weights  [cout][k]
  __shared__ _Float16 Bs[2][64][40];   // im2col   [pixel][k]

  const int w0 = blockIdx.x * 64;
  const int h  = blockIdx.y;
  const int b  = blockIdx.z;

  const int t    = threadIdx.x;
  const int lane = t & 31;
  const int wave = t >> 5;
  const int mt   = wave & 3;         // cout tile 0..3
  const int nt0  = (wave >> 2) << 1; // pixel tiles nt0, nt0+1

  const int frow = t >> 2;           // 0..63 (row / pixel for staging)
  const int fk   = (t & 3) << 3;     // 0,8,16,24

  v8f acc0 = {};
  v8f acc1 = {};

  auto stageA = [&](int buf, int ks) {
    const _Float16* src = wgt + (size_t)frow * KTOT + ks * 32 + fk;
#if USE_ASYNC_LDS
    __builtin_amdgcn_global_load_async_to_lds_b128(gcast(src),
                                                   lcast(&As[buf][frow][fk]), 0, 0);
#else
    *(v8h*)&As[buf][frow][fk] = *(const v8h*)src;
#endif
  };

  auto stageB = [&](int buf, int ks) {
    const int kg  = ks * 32;
    const int tap = kg / CIN;
    const int cin = (kg % CIN) + fk;
    const int hin = h + tap / 3 - 1;
    const int win = w0 + frow + tap % 3 - 1;
    const _Float16* src = zbuf;
    if (hin >= 0 && hin < HH && win >= 0 && win < WW) {
      int c = cin;
      const _Float16* base = in0;
      if constexpr (CIN == 128) {
        if (c >= 64) { base = in1; c -= 64; }
      }
      src = base + ((size_t)((b * HH + hin) * WW + win) << 6) + c;
    }
#if USE_ASYNC_LDS
    __builtin_amdgcn_global_load_async_to_lds_b128(gcast(src),
                                                   lcast(&Bs[buf][frow][fk]), 0, 0);
#else
    *(v8h*)&Bs[buf][frow][fk] = *(const v8h*)src;
#endif
  };

  stageA(0, 0);
  stageB(0, 0);
  async_stage_wait();
  __syncthreads();

  for (int ks = 0; ks < NSTEP; ++ks) {
    const int cb = ks & 1;
    const int nb = cb ^ 1;
    if (ks + 1 < NSTEP) {   // prefetch next panels while this step computes
      stageA(nb, ks + 1);
      stageB(nb, ks + 1);
    }

    // A fragment (16x32 f16): lanes 0-15 hold K{0..15}, 16-31 K{16..31}
    const int arow = mt * 16 + (lane & 15);
    const int akl  = (lane < 16) ? 0 : 16;
    v8h alo = *(const v8h*)&As[cb][arow][akl];
    v8h ahi = *(const v8h*)&As[cb][arow][akl + 8];
    v16h afrag;
#pragma unroll
    for (int i = 0; i < 8; ++i) { afrag[i] = alo[i]; afrag[i + 8] = ahi[i]; }

    // B fragments (32x16 f16): lane = column, 16 contiguous K values
    const int bkl = (lane < 16) ? 0 : 16;
    const int bc0 = nt0 * 16 + (lane & 15);
    const int bc1 = bc0 + 16;
    v8h b0l = *(const v8h*)&Bs[cb][bc0][bkl];
    v8h b0h = *(const v8h*)&Bs[cb][bc0][bkl + 8];
    v8h b1l = *(const v8h*)&Bs[cb][bc1][bkl];
    v8h b1h = *(const v8h*)&Bs[cb][bc1][bkl + 8];
    v16h bf0, bf1;
#pragma unroll
    for (int i = 0; i < 8; ++i) {
      bf0[i] = b0l[i]; bf0[i + 8] = b0h[i];
      bf1[i] = b1l[i]; bf1[i + 8] = b1h[i];
    }

    acc0 = __builtin_amdgcn_wmma_f32_16x16x32_f16(false, afrag, false, bf0,
                                                  (short)0, acc0, false, false);
    acc1 = __builtin_amdgcn_wmma_f32_16x16x32_f16(false, afrag, false, bf1,
                                                  (short)0, acc1, false, false);

    async_stage_wait();
    __syncthreads();
  }

  // epilogue: lane holds 8 *contiguous* couts (cbase..cbase+7) per tile ->
  // pack to v8h, single b128 NHWC store per tile.
  const float a = PRELU ? alpha[0] : 0.f;
  const int mhi   = (lane < 16) ? 0 : 8;
  const int pxb   = lane & 15;
  const int cbase = mt * 16 + mhi;
  v8h o0, o1;
#pragma unroll
  for (int r = 0; r < 8; ++r) {
    const float bv = bias[cbase + r];
    float v0 = acc0[r] + bv;
    float v1 = acc1[r] + bv;
    if (PRELU) {
      v0 = v0 >= 0.f ? v0 : a * v0;
      v1 = v1 >= 0.f ? v1 : a * v1;
    }
    o0[r] = (_Float16)v0;
    o1[r] = (_Float16)v1;
  }
  const size_t rowbase = (size_t)((b * HH + h) * WW + w0) << 6;
  *(v8h*)(out + rowbase + ((size_t)(nt0 * 16 + pxb) << 6) + cbase) = o0;
  *(v8h*)(out + rowbase + ((size_t)((nt0 + 1) * 16 + pxb) << 6) + cbase) = o1;
}

// ---------------------------------------------------------------------------
// t5: 3x3 conv 64->1, bias, no activation. NHWC f16 -> fp32 output plane.
// ---------------------------------------------------------------------------
__global__ void conv_t5_kernel(const _Float16* __restrict__ in,  // [8,256,256,64]
                               const _Float16* __restrict__ w,   // [9][64] f16
                               const float* __restrict__ bias,   // [1]
                               float* __restrict__ out) {        // [8,256,256]
  int i = blockIdx.x * blockDim.x + threadIdx.x;
  if (i >= 8 * 65536) return;
  int b = i >> 16;
  int h = (i >> 8) & 255;
  int wx = i & 255;
  float acc = bias[0];
#pragma unroll
  for (int tap = 0; tap < 9; ++tap) {
    int hh = h + tap / 3 - 1, ww = wx + tap % 3 - 1;
    if (hh < 0 || hh >= HH || ww < 0 || ww >= WW) continue;
    const _Float16* ip = in + ((size_t)((b * HH + hh) * WW + ww) << 6);
    const _Float16* wp = w + tap * 64;
#pragma unroll
    for (int c = 0; c < 64; c += 8) {
      v8h iv = *(const v8h*)(ip + c);
      v8h wv = *(const v8h*)(wp + c);
#pragma unroll
      for (int j = 0; j < 8; ++j) acc += (float)iv[j] * (float)wv[j];
    }
  }
  out[i] = acc;
}

// ---------------------------------------------------------------------------
// Host orchestration
// ---------------------------------------------------------------------------
extern "C" void kernel_launch(void* const* d_in, const int* in_sizes, int n_in,
                              void* d_out, int out_size, void* d_ws, size_t ws_size,
                              hipStream_t stream) {
  const float* x        = (const float*)d_in[0];
  const float* sample_w = (const float*)d_in[1];
  const float* up_w     = (const float*)d_in[2];
  const float* up_b     = (const float*)d_in[3];
  const float* conv1_w  = (const float*)d_in[4];
  const float* conv1_b  = (const float*)d_in[5];
  const float* conv1_a  = (const float*)d_in[6];
  const float* fus_w    = (const float*)d_in[7];
  const float* fus_b    = (const float*)d_in[8];
  const float* fus_a    = (const float*)d_in[9];
  const float* t2_w     = (const float*)d_in[10];
  const float* t2_b     = (const float*)d_in[11];
  const float* t2_a     = (const float*)d_in[12];
  const float* t3_w     = (const float*)d_in[13];
  const float* t3_b     = (const float*)d_in[14];
  const float* t3_a     = (const float*)d_in[15];
  const float* t4_w     = (const float*)d_in[16];
  const float* t4_b     = (const float*)d_in[17];
  const float* t4_a     = (const float*)d_in[18];
  const float* t5_w     = (const float*)d_in[19];
  const float* t5_b     = (const float*)d_in[20];
  float* out = (float*)d_out;
  (void)in_sizes; (void)n_in; (void)out_size; (void)ws_size;

  char* ws = (char*)d_ws;
  size_t off = 0;
  auto walloc = [&](size_t bytes) -> void* {
    void* p = ws + off;
    off = (off + bytes + 255) & ~(size_t)255;
    return p;
  };

  _Float16* t2f  = (_Float16*)walloc((size_t)512 * 576 * 2);
  _Float16* t3f  = (_Float16*)walloc((size_t)512 * 576 * 2);
  _Float16* t4f  = (_Float16*)walloc((size_t)512 * 576 * 2);
  _Float16* fusf = (_Float16*)walloc((size_t)448 * 1152 * 2);
  _Float16* t5f  = (_Float16*)walloc((size_t)8 * 576 * 2);
  _Float16* zbuf = (_Float16*)walloc((size_t)128 * 2);
  float*    samp = (float*)walloc((size_t)8 * 13 * 64 * 4);
  _Float16* img  = (_Float16*)walloc((size_t)8 * 65536 * 2);
  const size_t ACT_BYTES = (size_t)8 * 256 * 256 * 64 * 2;   // 67 MB each
  _Float16* CUR  = (_Float16*)walloc(ACT_BYTES);
  _Float16* HEAD = (_Float16*)walloc(ACT_BYTES);
  _Float16* X1   = (_Float16*)walloc(ACT_BYTES);
  _Float16* X2   = (_Float16*)walloc(ACT_BYTES);

  auto cdiv = [](int a, int b) { return (a + b - 1) / b; };

  zero_init_kernel<<<1, 128, 0, stream>>>(zbuf, 128);

  // weight repack (f32 -> f16, tap-major K)
  convert_w_kernel<<<cdiv(512 * 64 * 9, 256), 256, 0, stream>>>(t2_w, t2f, 512, 64);
  convert_w_kernel<<<cdiv(512 * 64 * 9, 256), 256, 0, stream>>>(t3_w, t3f, 512, 64);
  convert_w_kernel<<<cdiv(512 * 64 * 9, 256), 256, 0, stream>>>(t4_w, t4f, 512, 64);
  convert_w_kernel<<<cdiv(448 * 128 * 9, 256), 256, 0, stream>>>(fus_w, fusf, 448, 128);
  convert_w_kernel<<<cdiv(8 * 64 * 9, 256), 256, 0, stream>>>(t5_w, t5f, 8, 64);

  dim3 cgrid(4, 256, 8);  // (W/64, H, B)

  auto head = [&](int g, _Float16* dst) {
    sample_kernel<<<cdiv(8 * 13 * 64, 256), 256, 0, stream>>>(
        x, sample_w + (size_t)g * 13 * 1024, samp);
    up_dts_kernel<<<cdiv(8 * 65536, 256), 256, 0, stream>>>(
        samp, up_w + (size_t)g * 1024 * 13, up_b + g * 1024, img);
    conv1_kernel<<<cdiv(8 * 65536, 256), 256, 0, stream>>>(
        img, conv1_w + (size_t)g * 64 * 9, conv1_b + g * 64, conv1_a + g, dst);
  };

  auto tail = [&](int g, const _Float16* src) {
    conv3x3_wmma_kernel<64, true><<<cgrid, 256, 0, stream>>>(
        src, nullptr, t2f + (size_t)g * 64 * 576, t2_b + g * 64, t2_a + g, zbuf, X1);
    conv3x3_wmma_kernel<64, true><<<cgrid, 256, 0, stream>>>(
        X1, nullptr, t3f + (size_t)g * 64 * 576, t3_b + g * 64, t3_a + g, zbuf, X2);
    conv3x3_wmma_kernel<64, true><<<cgrid, 256, 0, stream>>>(
        X2, nullptr, t4f + (size_t)g * 64 * 576, t4_b + g * 64, t4_a + g, zbuf, X1);
    conv_t5_kernel<<<cdiv(8 * 65536, 256), 256, 0, stream>>>(
        X1, t5f + (size_t)g * 576, t5_b + g, out + (size_t)g * 8 * 65536);
  };

  head(0, CUR);
  tail(0, CUR);
  for (int m = 0; m < 7; ++m) {
    head(m + 1, HEAD);
    // fusion: concat([cur, head]) -> 128ch conv -> 64ch, implicit concat via dual source
    conv3x3_wmma_kernel<128, true><<<cgrid, 256, 0, stream>>>(
        CUR, HEAD, fusf + (size_t)m * 64 * 1152, fus_b + m * 64, fus_a + m, zbuf, X2);
    _Float16* tmp = CUR; CUR = X2; X2 = tmp;
    tail(m + 1, CUR);
  }
}
